// MoEBlock_35656818492150
// MI455X (gfx1250) — matrix operand order repaired and checked by reading, not compiled, not run on hardware
//
#include <hip/hip_runtime.h>
#include <hip/hip_bf16.h>
#include <math.h>

#define NE   4
#define DIM  384
#define HID  1536
#define MT   64      // tokens per block
#define KC   64      // HID chunk
#define XSS  392     // x LDS row stride (bf16 elems), padded for bank-conflict-free A reads
#define HSS  72      // h LDS row stride

// sched_barrier mask: VALU|SALU|DS(all/read/write)|transcendental may cross;
// VMEM and WMMA may NOT cross -> enforces prefetch distance.
#define SB_MASK 0x786

typedef __attribute__((ext_vector_type(16))) __bf16 v16bf;
typedef __attribute__((ext_vector_type(8)))  __bf16 v8bf;
typedef __attribute__((ext_vector_type(8)))  float  v8f;

__device__ __forceinline__ float gelu_exact(float v) {
    return 0.5f * v * (1.0f + erff(v * 0.70710678118654752440f));
}

// Build a 16x32 bf16 A-fragment for one lane from row-major LDS:
// lane<16  holds row (lane),  K = {0..15}   (klo = 0)
// lane>=16 holds same row,    K = {16..31}  (klo = 16, baked into ptr)
__device__ __forceinline__ v16bf make_a(const __bf16* p) {
    union { v16bf v; v8bf h[2]; } u;
    u.h[0] = *(const v8bf*)(p);
    u.h[1] = *(const v8bf*)(p + 8);
    return u.v;
}

// ---------------------------------------------------------------------------
// Transpose + f32 -> bf16 convert:  src[e][K][N]  ->  dst[e][N][K]
// ---------------------------------------------------------------------------
__global__ __launch_bounds__(256)
void transpose_to_bf16(const float* __restrict__ src, __bf16* __restrict__ dst,
                       int K, int N)
{
    __shared__ float tile[32][33];
    const int e  = blockIdx.z;
    const int n0 = blockIdx.x * 32;
    const int k0 = blockIdx.y * 32;
    const int tx = threadIdx.x, ty = threadIdx.y;
    const float* s = src + (size_t)e * K * N;
    __bf16* d = dst + (size_t)e * K * N;
    #pragma unroll
    for (int r = 0; r < 4; ++r)
        tile[ty + r * 8][tx] = s[(size_t)(k0 + ty + r * 8) * N + n0 + tx];
    __syncthreads();
    #pragma unroll
    for (int r = 0; r < 4; ++r)
        d[(size_t)(n0 + ty + r * 8) * K + k0 + tx] = (__bf16)tile[tx][ty + r * 8];
}

// ---------------------------------------------------------------------------
// Gating: softmax over 4 experts, top-2, renormalize. One wave per token.
// ---------------------------------------------------------------------------
__global__ __launch_bounds__(128)
void gate_kernel(const float* __restrict__ x, const float* __restrict__ gw,
                 const float* __restrict__ gb, float* __restrict__ gout, int T)
{
    const int lane = threadIdx.x & 31;
    const int t = blockIdx.x * 4 + (threadIdx.x >> 5);
    if (t >= T) return;
    const float* xr = x + (size_t)t * DIM;
    float a[NE] = {0.f, 0.f, 0.f, 0.f};
    for (int d = lane; d < DIM; d += 32) {
        const float xv = xr[d];
        #pragma unroll
        for (int e = 0; e < NE; ++e) a[e] += xv * gw[d * NE + e];
    }
    #pragma unroll
    for (int off = 16; off > 0; off >>= 1) {
        #pragma unroll
        for (int e = 0; e < NE; ++e) a[e] += __shfl_xor(a[e], off, 32);
    }
    if (lane == 0) {
        float s[NE], p[NE];
        #pragma unroll
        for (int e = 0; e < NE; ++e) s[e] = a[e] + gb[e];
        float m = s[0];
        #pragma unroll
        for (int e = 1; e < NE; ++e) m = fmaxf(m, s[e]);
        float tot = 0.f;
        #pragma unroll
        for (int e = 0; e < NE; ++e) { p[e] = expf(s[e] - m); tot += p[e]; }
        #pragma unroll
        for (int e = 0; e < NE; ++e) p[e] /= tot;
        int i0 = 0;
        #pragma unroll
        for (int e = 1; e < NE; ++e) if (p[e] > p[i0]) i0 = e;       // first-max on ties
        int i1 = (i0 == 0) ? 1 : 0;
        #pragma unroll
        for (int e = 0; e < NE; ++e) if (e != i0 && p[e] > p[i1]) i1 = e;
        const float den = p[i0] + p[i1] + 1e-9f;
        #pragma unroll
        for (int e = 0; e < NE; ++e)
            gout[(size_t)t * NE + e] = (e == i0 || e == i1) ? p[e] / den : 0.f;
    }
}

// ---------------------------------------------------------------------------
// Fused MoE FFN: out[t] = sum_e s[t,e] * ( gelu(x[t]@w1[e]+b1[e]) @ w2[e] + b2[e] )
// Block = 256 threads = 8 waves, 64 tokens.
// Wave footprint = 2 row-strips x N columns (B fragment shared by 2 WMMAs).
// B loads run a distance-2 software pipeline enforced by sched_barriers;
// first-use B tiles are prefetched across the workgroup barrier.
// ---------------------------------------------------------------------------
__global__ __launch_bounds__(256, 1)
void moe_ffn_kernel(const float* __restrict__ x,
                    const float* __restrict__ gate,   // [T][NE]
                    const __bf16* __restrict__ w1t,   // [NE][HID][DIM]
                    const float* __restrict__ b1,     // [NE][HID]
                    const __bf16* __restrict__ w2t,   // [NE][DIM][HID]
                    const float* __restrict__ b2,     // [NE][DIM]
                    float* __restrict__ out)          // [T][DIM]
{
    __shared__ __align__(16) __bf16 xs[MT * XSS];   // 50176 B
    __shared__ __align__(16) __bf16 hs[MT * HSS];   //  9216 B
    __shared__ float sg[MT * NE];                   //  1024 B

    const int t0   = blockIdx.x * MT;
    const int tid  = threadIdx.x;
    const int lane = tid & 31;
    const int wave = tid >> 5;        // 0..7
    const int sp   = wave & 1;        // strip pair: strips {2sp, 2sp+1}
    const int s0   = 2 * sp;
    const int cg   = wave >> 1;       // 0..3: column / column-group index
    const int col  = lane & 15;       // N within a 16x16 tile (B/C/D layout)
    const int hi   = lane >> 4;       // lane half
    const int klo  = hi * 16;         // A-fragment K sub-offset
    const int bk   = hi * 16;         // B-fragment K sub-offset
    const int rb0  = s0 * 16 + hi * 8;        // C/D row base, strip s0
    const int rb1  = rb0 + 16;                // C/D row base, strip s0+1

    // Stage x tile (f32 -> bf16) and gate weights into LDS.
    for (int i = tid; i < MT * DIM; i += 256) {
        int r = i / DIM, c = i - r * DIM;
        xs[r * XSS + c] = (__bf16)x[(size_t)(t0 + r) * DIM + c];
    }
    for (int i = tid; i < MT * NE; i += 256)
        sg[i] = gate[(size_t)t0 * NE + i];
    __syncthreads();

    v8f acc[12];                      // 6 cols x 2 strips of 16x16 per wave
    #pragma unroll
    for (int i = 0; i < 12; ++i) {
        #pragma unroll
        for (int j = 0; j < 8; ++j) acc[i][j] = 0.0f;
    }

    const __bf16* ap0  = &xs[(s0 * 16 + col) * XSS + klo];
    const __bf16* ap1  = ap0 + 16 * XSS;
    const __bf16* hap0 = &hs[(s0 * 16 + col) * HSS + klo];
    const __bf16* hap1 = hap0 + 16 * HSS;

    for (int e = 0; e < NE; ++e) {
        const __bf16* w1e = w1t + (size_t)e * HID * DIM;
        const __bf16* w2e = w2t + (size_t)e * DIM * HID;
        // Prefetch first GEMM1 B tile of this expert (kc = 0).
        v16bf g1b = *(const v16bf*)(w1e + (size_t)(cg * 16 + col) * DIM + bk);

        for (int kc = 0; kc < HID; kc += KC) {
            // ---- GEMM1: h(64xKC) = x @ w1 slice; wave = 2 strips x 1 col tile
            v8f h0, h1;
            #pragma unroll
            for (int j = 0; j < 8; ++j) { h0[j] = 0.f; h1[j] = 0.f; }
            const int n0 = kc + cg * 16 + col;           // w1 column (HID index)
            const __bf16* bp = w1e + (size_t)n0 * DIM + bk;
            v16bf bc = g1b;                               // B(k)
            v16bf bn = *(const v16bf*)(bp + 32);          // B(k+32)
            #pragma unroll
            for (int k = 0; k < DIM; k += 32) {
                v16bf b2n = bn;
                if (k + 64 < DIM) b2n = *(const v16bf*)(bp + k + 64);  // B(k+64)
                __builtin_amdgcn_sched_barrier(SB_MASK);  // keep 2 B tiles in flight
                v16bf a0 = make_a(ap0 + k);
                v16bf a1 = make_a(ap1 + k);
                h0 = __builtin_amdgcn_wmma_f32_16x16x32_bf16(false, a0, false, bc,
                                                             (short)0, h0, false, false);
                h1 = __builtin_amdgcn_wmma_f32_16x16x32_bf16(false, a1, false, bc,
                                                             (short)0, h1, false, false);
                bc = bn; bn = b2n;
            }

            // Prefetch (pinned above the GELU block by a sched_barrier so the
            // VALU work + workgroup barrier hide their latency):
            //  - next chunk's first GEMM1 B tile
            //  - this chunk's GEMM2 B tiles for j=0 and j=1
            if (kc + KC < HID)
                g1b = *(const v16bf*)(w1e + (size_t)(n0 + KC) * DIM + bk);
            const __bf16* wp0 = w2e + (size_t)((cg * 6) * 16 + col) * HID + kc + bk;
            v16bf c0 = *(const v16bf*)(wp0);
            v16bf c1 = *(const v16bf*)(wp0 + 32);
            v16bf d0 = *(const v16bf*)(wp0 + (size_t)16 * HID);
            v16bf d1 = *(const v16bf*)(wp0 + (size_t)16 * HID + 32);
            __builtin_amdgcn_sched_barrier(SB_MASK);

            // bias + exact gelu + fold per-token gate weight, stage to LDS (bf16)
            const float bias = b1[e * HID + n0];
            const int   hc   = cg * 16 + col;            // column within hs
            #pragma unroll
            for (int r = 0; r < 8; ++r) {
                const float sv0 = sg[(rb0 + r) * NE + e];
                const float sv1 = sg[(rb1 + r) * NE + e];
                hs[(rb0 + r) * HSS + hc] = (__bf16)(gelu_exact(h0[r] + bias) * sv0);
                hs[(rb1 + r) * HSS + hc] = (__bf16)(gelu_exact(h1[r] + bias) * sv1);
            }
            __syncthreads();

            // ---- GEMM2: acc += h @ w2 slice; wave = 2 strips x 6 col tiles
            const v16bf ha00 = make_a(hap0);
            const v16bf ha01 = make_a(hap0 + 32);
            const v16bf ha10 = make_a(hap1);
            const v16bf ha11 = make_a(hap1 + 32);
            #pragma unroll
            for (int j = 0; j < 6; ++j) {
                v16bf e0 = d0, e1 = d1;
                if (j + 2 < 6) {                          // prefetch column tile j+2
                    const __bf16* wpn = wp0 + (size_t)(j + 2) * 16 * HID;
                    e0 = *(const v16bf*)(wpn);
                    e1 = *(const v16bf*)(wpn + 32);
                }
                __builtin_amdgcn_sched_barrier(SB_MASK);  // keep 2 B pairs in flight
                acc[2*j]   = __builtin_amdgcn_wmma_f32_16x16x32_bf16(false, ha00, false, c0,
                                                                     (short)0, acc[2*j], false, false);
                acc[2*j+1] = __builtin_amdgcn_wmma_f32_16x16x32_bf16(false, ha10, false, c0,
                                                                     (short)0, acc[2*j+1], false, false);
                acc[2*j]   = __builtin_amdgcn_wmma_f32_16x16x32_bf16(false, ha01, false, c1,
                                                                     (short)0, acc[2*j], false, false);
                acc[2*j+1] = __builtin_amdgcn_wmma_f32_16x16x32_bf16(false, ha11, false, c1,
                                                                     (short)0, acc[2*j+1], false, false);
                c0 = d0; c1 = d1; d0 = e0; d1 = e1;
            }
            __syncthreads();   // hs reused next chunk
        }
    }

    // Epilogue: add sum_e s[t,e]*b2[e][d]; single deterministic store per element.
    #pragma unroll
    for (int j = 0; j < 6; ++j) {
        const int n = (cg * 6 + j) * 16 + col;
        float b2v[NE];
        #pragma unroll
        for (int e2 = 0; e2 < NE; ++e2) b2v[e2] = b2[e2 * DIM + n];
        #pragma unroll
        for (int r = 0; r < 8; ++r) {
            const int row0 = rb0 + r;
            const int row1 = rb1 + r;
            float ba0 = 0.f, ba1 = 0.f;
            #pragma unroll
            for (int e2 = 0; e2 < NE; ++e2) {
                ba0 += sg[row0 * NE + e2] * b2v[e2];
                ba1 += sg[row1 * NE + e2] * b2v[e2];
            }
            out[(size_t)(t0 + row0) * DIM + n] = acc[2*j][r]   + ba0;
            out[(size_t)(t0 + row1) * DIM + n] = acc[2*j+1][r] + ba1;
        }
    }
}

// ---------------------------------------------------------------------------
extern "C" void kernel_launch(void* const* d_in, const int* in_sizes, int n_in,
                              void* d_out, int out_size, void* d_ws, size_t ws_size,
                              hipStream_t stream)
{
    const float* x  = (const float*)d_in[0];
    const float* gw = (const float*)d_in[1];
    const float* gb = (const float*)d_in[2];
    const float* w1 = (const float*)d_in[3];
    const float* b1 = (const float*)d_in[4];
    const float* w2 = (const float*)d_in[5];
    const float* b2 = (const float*)d_in[6];
    float* out = (float*)d_out;
    const int T = in_sizes[0] / DIM;   // B*S = 32768

    // Workspace carve-up (~10 MB): gate[T][4] f32, w1t/w2t bf16 transposed.
    char* ws = (char*)d_ws;
    float* gate = (float*)ws;
    size_t off = ((size_t)T * NE * sizeof(float) + 255) & ~(size_t)255;
    __bf16* w1t = (__bf16*)(ws + off);
    off += ((size_t)NE * DIM * HID * sizeof(__bf16) + 255) & ~(size_t)255;
    __bf16* w2t = (__bf16*)(ws + off);

    dim3 tb(32, 8);
    transpose_to_bf16<<<dim3(HID / 32, DIM / 32, NE), tb, 0, stream>>>(w1, w1t, DIM, HID);
    transpose_to_bf16<<<dim3(DIM / 32, HID / 32, NE), tb, 0, stream>>>(w2, w2t, HID, DIM);
    gate_kernel<<<(T + 3) / 4, 128, 0, stream>>>(x, gw, gb, gate, T);
    moe_ffn_kernel<<<T / MT, 256, 0, stream>>>(x, gate, w1t, b1, w2t, b2, out);
}
